// Model_55396488184355
// MI455X (gfx1250) — matrix operand order repaired and checked by reading, not compiled, not run on hardware
//
#include <hip/hip_runtime.h>

typedef __attribute__((ext_vector_type(16))) _Float16 v16h;
typedef __attribute__((ext_vector_type(8)))  float    v8f;
typedef __attribute__((ext_vector_type(4)))  int      v4i;

#define NSAMP 32768
#define DM    128
#define TFR   128
#define FBK   512
#define XPW   (NSAMP + 1024)   // padded x row: 512 zeros front, 512 back

// ---- WMMA fragment index maps (CDNA5 ISA layouts, wave32) ----
// A-matrix 16x32 f16: lane m=lane&15, g=lane>>4; half h -> K
__device__ __forceinline__ int kmapA(int g, int h) {
  return ((h & 8) << 1) + g * 8 + (h & 7);
}
// B-matrix 32x16 f16: lane n=lane&15, g=lane>>4; half h -> K (same as A)
__device__ __forceinline__ int kmapB(int g, int h) {
  return ((h & 8) << 1) + g * 8 + (h & 7);
}

__device__ __forceinline__ v8f wmma_f16(v16h a, v16h b, v8f c) {
  return __builtin_amdgcn_wmma_f32_16x16x32_f16(false, a, false, b, (short)0, c,
                                                false, false);
}

// ---------------------------------------------------------------- cast fb
__global__ void k_cast_fb(const float* __restrict__ fb, _Float16* __restrict__ fb16,
                          int n) {
  int i = blockIdx.x * blockDim.x + threadIdx.x;
  if (i < n) fb16[i] = (_Float16)fb[i];
}

// ------------------------------------------------ zero-padded f16 copy of x
__global__ void k_cast_xpad(const float* __restrict__ x, _Float16* __restrict__ xpad) {
  int i = blockIdx.x * blockDim.x + threadIdx.x;
  if (i < 32 * XPW) {
    int b = i / XPW;
    int o = i - b * XPW;
    int src = o - 512;
    float v = (src >= 0 && src < NSAMP) ? x[(size_t)b * NSAMP + src] : 0.0f;
    xpad[i] = (_Float16)v;
  }
}

// --------------------------------------------- filterbank conv + abs + pool
// block = (batch, frame). 8 waves; wave w owns channels [16w,16w+16).
// frame f sums |conv(t)| for t in [f*256-256, f*256+256), /512.
// x window = xpad[b][f*256 .. f*256+1024) staged async into LDS.
__global__ __launch_bounds__(256) void k_conv_pool(
    const _Float16* __restrict__ xpad, const _Float16* __restrict__ fb16,
    float* __restrict__ frames) {
  __shared__ alignas(16) _Float16 sx[1024];
  int b = blockIdx.x >> 7;
  int f = blockIdx.x & 127;
  int tid = threadIdx.x;
  const _Float16* xw = xpad + (size_t)b * XPW + f * 256;

#if __has_builtin(__builtin_amdgcn_global_load_async_to_lds_b128)
  if (tid < 128) {  // 128 lanes x 16B = 2KB window, VGPR-bypassing async DMA
    typedef __attribute__((address_space(1))) v4i* gp4;
    typedef __attribute__((address_space(3))) v4i* lp4;
    __builtin_amdgcn_global_load_async_to_lds_b128(
        (gp4)(xw + tid * 8), (lp4)(sx + tid * 8), 0, 0);
  }
  asm volatile("s_wait_asynccnt 0" ::: "memory");
#else
  for (int i = tid; i < 1024; i += 256) sx[i] = xw[i];
#endif
  __syncthreads();

  int wave = tid >> 5;
  int lane = tid & 31;
  int n = lane & 15;
  int g = lane >> 4;
  int ct = wave;
  const _Float16* fbrow = fb16 + (size_t)(ct * 16 + n) * FBK;

  float partial[8] = {0.f, 0.f, 0.f, 0.f, 0.f, 0.f, 0.f, 0.f};
  for (int pt = 0; pt < 32; ++pt) {
    int tpos = f * 256 - 256 + pt * 16 + n;
    bool valid = (tpos >= 0) && (tpos < NSAMP);  // pooling zero-pad mask
    v8f acc = {};
    for (int ks = 0; ks < 16; ++ks) {
      v16h av, bv;
#pragma unroll
      for (int h = 0; h < 16; ++h) av[h] = fbrow[ks * 32 + kmapA(g, h)];
#pragma unroll
      for (int h = 0; h < 16; ++h)
        bv[h] = sx[pt * 16 + n + ks * 32 + kmapB(g, h)];
      acc = wmma_f16(av, bv, acc);
    }
    if (valid) {
#pragma unroll
      for (int r = 0; r < 8; ++r) partial[r] += fabsf(acc[r]);
    }
  }
#pragma unroll
  for (int r = 0; r < 8; ++r) {
    partial[r] += __shfl_xor(partial[r], 1, 32);
    partial[r] += __shfl_xor(partial[r], 2, 32);
    partial[r] += __shfl_xor(partial[r], 4, 32);
    partial[r] += __shfl_xor(partial[r], 8, 32);
  }
  if (n == 0) {
    int cbase = ct * 16 + g * 8;
#pragma unroll
    for (int r = 0; r < 8; ++r)
      frames[((size_t)b * TFR + f) * DM + cbase + r] =
          partial[r] * (1.0f / (float)FBK);
  }
}

// --------------------------------------------------------- embed + pos-enc
__global__ __launch_bounds__(128) void k_embed(
    const float* __restrict__ frames, const float* __restrict__ W,
    const float* __restrict__ bias, float* __restrict__ h) {
  int row = blockIdx.x;  // b*128 + t
  int t = row & 127;
  int o = threadIdx.x;
  const float* wr = W + (size_t)o * 161;
  const float* fr = frames + (size_t)row * DM;
  float acc = bias[o];
  for (int c = 0; c < DM; ++c) acc += fr[c] * wr[c];
  float pos = -1.0f + 2.0f * (float)t / 127.0f;
  acc += pos * wr[128];
  float fq = 3.14159265358979323846f;
#pragma unroll
  for (int i = 0; i < 16; ++i) {
    float ang = pos * fq;
    acc += __sinf(ang) * wr[129 + i] + __cosf(ang) * wr[145 + i];
    fq *= 2.0f;
  }
  h[(size_t)row * DM + o] = acc;
}

// -------------------- C = A @ W^T + b, K=128, WMMA, A-fragments cached
// one wave per block; grid = M/16; wave loops all N-tiles reusing A regs
__global__ __launch_bounds__(32) void k_gemm128(
    const float* __restrict__ A, const float* __restrict__ W,
    const float* __restrict__ bias, float* __restrict__ C,
    int N, int relu) {
  int mt = blockIdx.x;
  int lane = threadIdx.x;
  int n = lane & 15, g = lane >> 4;
  const float* arow = A + (size_t)(mt * 16 + n) * 128;
  v16h a0, a1, a2, a3;
#pragma unroll
  for (int h = 0; h < 16; ++h) {
    a0[h] = (_Float16)arow[0 + kmapA(g, h)];
    a1[h] = (_Float16)arow[32 + kmapA(g, h)];
    a2[h] = (_Float16)arow[64 + kmapA(g, h)];
    a3[h] = (_Float16)arow[96 + kmapA(g, h)];
  }
  int ntiles = N >> 4;
  for (int nt = 0; nt < ntiles; ++nt) {
    const float* wrow = W + (size_t)(nt * 16 + n) * 128;
    int pn = (nt + 1 < ntiles) ? nt + 1 : nt;
    __builtin_prefetch(W + (size_t)(pn * 16 + n) * 128, 0, 1);
    v8f acc = {};
    v16h bv;
#pragma unroll
    for (int h = 0; h < 16; ++h) bv[h] = (_Float16)wrow[0 + kmapB(g, h)];
    acc = wmma_f16(a0, bv, acc);
#pragma unroll
    for (int h = 0; h < 16; ++h) bv[h] = (_Float16)wrow[32 + kmapB(g, h)];
    acc = wmma_f16(a1, bv, acc);
#pragma unroll
    for (int h = 0; h < 16; ++h) bv[h] = (_Float16)wrow[64 + kmapB(g, h)];
    acc = wmma_f16(a2, bv, acc);
#pragma unroll
    for (int h = 0; h < 16; ++h) bv[h] = (_Float16)wrow[96 + kmapB(g, h)];
    acc = wmma_f16(a3, bv, acc);
    int col = nt * 16 + n;
    float bb = bias[col];
#pragma unroll
    for (int r = 0; r < 8; ++r) {
      float v = acc[r] + bb;
      if (relu) v = fmaxf(v, 0.0f);
      C[(size_t)(mt * 16 + g * 8 + r) * N + col] = v;
    }
  }
}

// ------------------------------------------- attention per (batch, head)
__global__ __launch_bounds__(256) void k_attn(
    const float* __restrict__ qkv, float* __restrict__ attn_o) {
  __shared__ float ss[TFR * TFR];        // 64KB: scores -> probs in place
  __shared__ _Float16 sv[TFR * 32];      // 8KB: V tile in f16
  int b = blockIdx.x >> 2;
  int hd = blockIdx.x & 3;
  int tid = threadIdx.x;
  int wave = tid >> 5, lane = tid & 31;
  int n = lane & 15, g = lane >> 4;
  const float* qb = qkv + (size_t)b * TFR * 384;

  // stage V (f32 -> f16) for the PV GEMM
  for (int i = tid; i < TFR * 32; i += 256) {
    int t = i >> 5, dcol = i & 31;
    sv[i] = (_Float16)qb[(size_t)t * 384 + 256 + hd * 32 + dcol];
  }

  {  // scores = q k^T / sqrt(32): K=32 -> one WMMA per 16x16 tile
    int qt = wave;
    v16h av;
#pragma unroll
    for (int h = 0; h < 16; ++h)
      av[h] = (_Float16)qb[(size_t)(qt * 16 + n) * 384 + hd * 32 + kmapA(g, h)];
    for (int kt = 0; kt < 8; ++kt) {
      v16h bv;
#pragma unroll
      for (int h = 0; h < 16; ++h)
        bv[h] = (_Float16)qb[(size_t)(kt * 16 + n) * 384 + 128 + hd * 32 +
                             kmapB(g, h)];
      v8f acc = {};
      acc = wmma_f16(av, bv, acc);
#pragma unroll
      for (int r = 0; r < 8; ++r)
        ss[(qt * 16 + g * 8 + r) * TFR + kt * 16 + n] =
            acc[r] * 0.17677669529663687f;
    }
  }
  __syncthreads();
  if (tid < TFR) {  // in-place f32 softmax, one row per thread (race-free)
    float mx = -3.0e38f;
    for (int i = 0; i < TFR; ++i) mx = fmaxf(mx, ss[tid * TFR + i]);
    float s = 0.0f;
    for (int i = 0; i < TFR; ++i) {
      float e = __expf(ss[tid * TFR + i] - mx);
      ss[tid * TFR + i] = e;
      s += e;
    }
    float inv = 1.0f / s;
    for (int i = 0; i < TFR; ++i) ss[tid * TFR + i] *= inv;
  }
  __syncthreads();
  {  // O = att @ v : M=128, N=32, K=128 (A from ss, B from sv, both LDS)
    int mt = wave;
    for (int nt = 0; nt < 2; ++nt) {
      v8f acc = {};
      for (int ks = 0; ks < 4; ++ks) {
        v16h av, bv;
#pragma unroll
        for (int h = 0; h < 16; ++h)
          av[h] = (_Float16)ss[(mt * 16 + n) * TFR + ks * 32 + kmapA(g, h)];
#pragma unroll
        for (int h = 0; h < 16; ++h)
          bv[h] = sv[(ks * 32 + kmapB(g, h)) * 32 + nt * 16 + n];
        acc = wmma_f16(av, bv, acc);
      }
#pragma unroll
      for (int r = 0; r < 8; ++r)
        attn_o[((size_t)b * TFR + mt * 16 + g * 8 + r) * DM + hd * 32 +
               nt * 16 + n] = acc[r];
    }
  }
}

// ------------------------------------------------ residual add + layernorm
__global__ __launch_bounds__(128) void k_add_ln(
    float* __restrict__ h, const float* __restrict__ a,
    const float* __restrict__ w, const float* __restrict__ bb) {
  __shared__ float red[128];
  int row = blockIdx.x;
  int tid = threadIdx.x;
  float v = h[(size_t)row * DM + tid] + a[(size_t)row * DM + tid];
  red[tid] = v;
  __syncthreads();
  for (int s = 64; s > 0; s >>= 1) {
    if (tid < s) red[tid] += red[tid + s];
    __syncthreads();
  }
  float mean = red[0] * (1.0f / DM);
  __syncthreads();
  float d = v - mean;
  red[tid] = d * d;
  __syncthreads();
  for (int s = 64; s > 0; s >>= 1) {
    if (tid < s) red[tid] += red[tid + s];
    __syncthreads();
  }
  float var = red[0] * (1.0f / DM);
  h[(size_t)row * DM + tid] = d * rsqrtf(var + 1e-5f) * w[tid] + bb[tid];
}

// ------------------------------------------- attn scores + top-16 -> vecs
__global__ __launch_bounds__(128) void k_topk(
    const float* __restrict__ h, const float* __restrict__ attn_w,
    const float* __restrict__ attn_b, float* __restrict__ vecs) {
  __shared__ float sc[TFR];
  __shared__ float rv[TFR];
  __shared__ int ri[TFR];
  int b = blockIdx.x;
  int tid = threadIdx.x;
  const float* hr = h + ((size_t)b * TFR + tid) * DM;
  float acc = attn_b[0];
  for (int c = 0; c < DM; ++c) acc += hr[c] * attn_w[c];
  sc[tid] = acc;
  __syncthreads();
  for (int it = 0; it < 16; ++it) {
    rv[tid] = sc[tid];
    ri[tid] = tid;
    __syncthreads();
    for (int s = 64; s > 0; s >>= 1) {
      if (tid < s) {
        if (rv[tid + s] > rv[tid]) { rv[tid] = rv[tid + s]; ri[tid] = ri[tid + s]; }
      }
      __syncthreads();
    }
    int idx = ri[0];
    float val = rv[0];
    vecs[((size_t)b * 16 + it) * DM + tid] =
        h[((size_t)b * TFR + idx) * DM + tid] * val;
    __syncthreads();
    if (tid == 0) sc[idx] = -3.0e38f;
    __syncthreads();
  }
}

// ----------------------- amp / atom-argmax / pos-argmax per selected token
__global__ __launch_bounds__(128) void k_heads(
    const float* __restrict__ vecs, const float* __restrict__ amp_w,
    const float* __restrict__ amp_b, const float* __restrict__ pos_w,
    const float* __restrict__ pos_b, const float* __restrict__ atom_w,
    const float* __restrict__ atom_b, float* __restrict__ ampv,
    int* __restrict__ aidx, int* __restrict__ pidx) {
  __shared__ float red[128];
  __shared__ float rv[128];
  __shared__ int ri[128];
  int bj = blockIdx.x;
  int tid = threadIdx.x;
  const float* vec = vecs + (size_t)bj * DM;

  red[tid] = vec[tid] * amp_w[tid];
  __syncthreads();
  for (int s = 64; s > 0; s >>= 1) { if (tid < s) red[tid] += red[tid + s]; __syncthreads(); }
  float av = fmaxf(red[0] + amp_b[0], 0.0f);

  float bestv = -3.0e38f; int besti = 0;
  for (int c = 0; c < 2; ++c) {
    int l = tid + c * 128;
    float lv = atom_b[l];
    const float* wr = atom_w + (size_t)l * DM;
    for (int k = 0; k < DM; ++k) lv += vec[k] * wr[k];
    if (lv > bestv) { bestv = lv; besti = l; }
  }
  rv[tid] = bestv; ri[tid] = besti;
  __syncthreads();
  for (int s = 64; s > 0; s >>= 1) {
    if (tid < s) {
      bool take = (rv[tid + s] > rv[tid]) ||
                  (rv[tid + s] == rv[tid] && ri[tid + s] < ri[tid]);
      if (take) { rv[tid] = rv[tid + s]; ri[tid] = ri[tid + s]; }
    }
    __syncthreads();
  }
  int atom_i = ri[0];
  __syncthreads();

  bestv = -3.0e38f; besti = 0;
  for (int c = 0; c < 4; ++c) {
    int l = tid + c * 128;
    float lv = pos_b[l];
    const float* wr = pos_w + (size_t)l * DM;
    for (int k = 0; k < DM; ++k) lv += vec[k] * wr[k];
    if (lv > bestv) { bestv = lv; besti = l; }
  }
  rv[tid] = bestv; ri[tid] = besti;
  __syncthreads();
  for (int s = 64; s > 0; s >>= 1) {
    if (tid < s) {
      bool take = (rv[tid + s] > rv[tid]) ||
                  (rv[tid + s] == rv[tid] && ri[tid + s] < ri[tid]);
      if (take) { rv[tid] = rv[tid + s]; ri[tid] = ri[tid + s]; }
    }
    __syncthreads();
  }
  if (tid == 0) {
    ampv[bj] = av;
    aidx[bj] = atom_i;
    pidx[bj] = ri[0];
  }
}

// ------------------------------------------------- zero + atom scatter-add
__global__ void k_zero(float* __restrict__ out, int nelem) {
  int i = blockIdx.x * blockDim.x + threadIdx.x;
  if (i < nelem) out[i] = 0.0f;
}

__global__ __launch_bounds__(256) void k_scatter(
    const float* __restrict__ dic, const float* __restrict__ ampv,
    const int* __restrict__ aidx, const int* __restrict__ pidx,
    float* __restrict__ out) {
  int b = blockIdx.x, j = blockIdx.y;
  int bj = b * 16 + j;
  int tid = threadIdx.x;
  int t = pidx[bj] * 64 + tid;  // repeat(.,64)+argmax -> shift = 64*argmax
  if (t < NSAMP) {
    float v = dic[(size_t)aidx[bj] * 256 + tid] * ampv[bj];
    atomicAdd(&out[(size_t)b * NSAMP + t], v);
  }
}

// --------------------------------------------------------------- launcher
extern "C" void kernel_launch(void* const* d_in, const int* in_sizes, int n_in,
                              void* d_out, int out_size, void* d_ws, size_t ws_size,
                              hipStream_t stream) {
  (void)in_sizes; (void)n_in; (void)out_size; (void)ws_size;
  const float* x        = (const float*)d_in[0];
  const float* fb       = (const float*)d_in[1];
  const float* embed_w  = (const float*)d_in[2];
  const float* embed_b  = (const float*)d_in[3];
  const float* in_proj_w= (const float*)d_in[4];
  const float* in_proj_b= (const float*)d_in[5];
  const float* out_w    = (const float*)d_in[6];
  const float* out_b    = (const float*)d_in[7];
  const float* lin1_w   = (const float*)d_in[8];
  const float* lin1_b   = (const float*)d_in[9];
  const float* lin2_w   = (const float*)d_in[10];
  const float* lin2_b   = (const float*)d_in[11];
  const float* ln1_w    = (const float*)d_in[12];
  const float* ln1_b    = (const float*)d_in[13];
  const float* ln2_w    = (const float*)d_in[14];
  const float* ln2_b    = (const float*)d_in[15];
  const float* attn_w   = (const float*)d_in[16];
  const float* attn_b   = (const float*)d_in[17];
  const float* amp_w    = (const float*)d_in[18];
  const float* amp_b    = (const float*)d_in[19];
  const float* pos_w    = (const float*)d_in[20];
  const float* pos_b    = (const float*)d_in[21];
  const float* atom_w   = (const float*)d_in[22];
  const float* atom_b   = (const float*)d_in[23];
  const float* dic      = (const float*)d_in[24];
  float* out = (float*)d_out;

  char* ws = (char*)d_ws;
  _Float16* fb16 = (_Float16*)ws; ws += (size_t)128 * 512 * 2;
  _Float16* xpad = (_Float16*)ws; ws += (size_t)32 * XPW * 2;
  float* frames  = (float*)ws;    ws += (size_t)32 * 128 * 128 * 4;
  float* hbuf    = (float*)ws;    ws += (size_t)4096 * 128 * 4;
  float* qkvb    = (float*)ws;    ws += (size_t)4096 * 384 * 4;
  float* attnb   = (float*)ws;    ws += (size_t)4096 * 128 * 4;
  float* tmpb    = (float*)ws;    ws += (size_t)4096 * 128 * 4;
  float* ffb     = (float*)ws;    ws += (size_t)4096 * 128 * 4;
  float* vecs    = (float*)ws;    ws += (size_t)32 * 16 * 128 * 4;
  float* ampv    = (float*)ws;    ws += (size_t)512 * 4;
  int*   aidx    = (int*)ws;      ws += (size_t)512 * 4;
  int*   pidx    = (int*)ws;      ws += (size_t)512 * 4;

  k_cast_fb<<<256, 256, 0, stream>>>(fb, fb16, 128 * 512);
  k_cast_xpad<<<(32 * XPW + 255) / 256, 256, 0, stream>>>(x, xpad);
  k_conv_pool<<<32 * 128, 256, 0, stream>>>(xpad, fb16, frames);
  k_embed<<<4096, 128, 0, stream>>>(frames, embed_w, embed_b, hbuf);
  for (int L = 0; L < 6; ++L) {
    k_gemm128<<<256, 32, 0, stream>>>(hbuf, in_proj_w + (size_t)L * 384 * 128,
                                      in_proj_b + L * 384, qkvb, 384, 0);
    k_attn<<<32 * 4, 256, 0, stream>>>(qkvb, attnb);
    k_gemm128<<<256, 32, 0, stream>>>(attnb, out_w + (size_t)L * 128 * 128,
                                      out_b + L * 128, tmpb, 128, 0);
    k_add_ln<<<4096, 128, 0, stream>>>(hbuf, tmpb, ln1_w + L * 128, ln1_b + L * 128);
    k_gemm128<<<256, 32, 0, stream>>>(hbuf, lin1_w + (size_t)L * 128 * 128,
                                      lin1_b + L * 128, ffb, 128, 1);
    k_gemm128<<<256, 32, 0, stream>>>(ffb, lin2_w + (size_t)L * 128 * 128,
                                      lin2_b + L * 128, tmpb, 128, 0);
    k_add_ln<<<4096, 128, 0, stream>>>(hbuf, tmpb, ln2_w + L * 128, ln2_b + L * 128);
  }
  k_topk<<<32, 128, 0, stream>>>(hbuf, attn_w, attn_b, vecs);
  k_heads<<<512, 128, 0, stream>>>(vecs, amp_w, amp_b, pos_w, pos_b, atom_w,
                                   atom_b, ampv, aidx, pidx);
  k_zero<<<(32 * NSAMP + 255) / 256, 256, 0, stream>>>(out, 32 * NSAMP);
  k_scatter<<<dim3(32, 16), 256, 0, stream>>>(dic, ampv, aidx, pidx, out);
}
